// Local_Feature_Extractor_67259187855876
// MI455X (gfx1250) — compile-verified, matrix operand order checked by reading, not run on hardware
//
#include <hip/hip_runtime.h>
#include <hip/hip_bf16.h>
#include <math.h>

// ---------------------------------------------------------------------------
// Problem constants (from reference)
// ---------------------------------------------------------------------------
#define BH     64          // B*H
#define TL_    256
#define AL_    2048
#define NMELS_ 80
#define HID_   256
#define TDIM_  512
#define NOUT   256         // every GEMM in this net has N == 256
#define BN_EPS 1e-5f

typedef __attribute__((ext_vector_type(16))) __bf16 v16bf;
typedef __attribute__((ext_vector_type(8)))  float  v8f;

// fp32 -> bf16 round-to-nearest-even (bit pattern)
__device__ __forceinline__ unsigned short f2bf(float f) {
  unsigned u = __float_as_uint(f);
  unsigned r = u + 0x7FFFu + ((u >> 16) & 1u);
  return (unsigned short)(r >> 16);
}

// ---------------------------------------------------------------------------
// Templated WMMA GEMM with on-the-fly im2col (conv taps), fused epilogue.
//   Out[bh, L, 256] = epilogue( A_conv[bh, L, KTAPS*CIN] @ B[KTAPS*CIN, 256] )
//   BMODE: 0 = conv weight W[N][CIN][KTAPS]
//          1 = batched   W[bh][KTOT][N]
//          2 = batched^T W[bh][N][KTOT]   (for q @ k^T)
//          3 = shared    W[KTOT][N]
// Tile 64x64 per workgroup (128 thr = 4 waves), wave = 2x2 of 16x16x32 WMMA.
// Double-buffered LDS staging so global loads for step k+1 overlap WMMA of k.
// ---------------------------------------------------------------------------
#define TM 64
#define TN 64
#define TK 32

template<int KTAPS, int CIN, int BMODE>
__device__ __forceinline__ void stage_tiles(
    unsigned short (*As)[TK + 4], unsigned short (*Bs)[TK + 4],
    const float* __restrict__ A, const float* __restrict__ W,
    int L, int batch, int pos0, int n0, int k0, int tid, bool prefetch)
{
  constexpr int KTOT = KTAPS * CIN;
  constexpr int PAD  = (KTAPS - 1) / 2;

  // ---- A tile: im2col gather, fp32 -> bf16 ----
  #pragma unroll
  for (int it = 0; it < (TM * TK) / 128; ++it) {
    int idx = tid + it * 128;
    int r  = idx >> 5;
    int kk = idx & 31;
    int kd = k0 + kk;
    float v = 0.0f;
    bool inK = (KTOT % TK == 0) || (kd < KTOT);
    if (inK) {
      int tap = (KTAPS == 1) ? 0 : (kd / CIN);       // compile-time divisor
      int ci  = (KTAPS == 1) ? kd : (kd - tap * CIN);
      int src = pos0 + r + tap - PAD;
      bool ok = (KTAPS == 1) || (src >= 0 && src < L);
      if (ok) {
        const float* ap = A + ((size_t)batch * L + src) * CIN + ci;
        v = *ap;
        if (prefetch) __builtin_prefetch(ap + TK, 0, 1);   // next k-step
      }
    }
    As[r][kk] = f2bf(v);
  }
  // ---- B tile, stored transposed: Bs[n][k] ----
  #pragma unroll
  for (int it = 0; it < (TN * TK) / 128; ++it) {
    int idx = tid + it * 128;
    int nn = idx >> 5;
    int kk = idx & 31;
    int kd = k0 + kk;
    float v = 0.0f;
    bool inK = (KTOT % TK == 0) || (kd < KTOT);
    if (inK) {
      int n = n0 + nn;
      if (BMODE == 0) {
        int tap = (KTAPS == 1) ? 0 : (kd / CIN);
        int ci  = (KTAPS == 1) ? kd : (kd - tap * CIN);
        v = W[((size_t)n * CIN + ci) * KTAPS + tap];
      } else if (BMODE == 3) {
        const float* wp = W + (size_t)kd * NOUT + n;
        v = *wp;
        if (prefetch) __builtin_prefetch(wp + (size_t)TK * NOUT, 0, 1);
      } else if (BMODE == 1) {
        v = W[((size_t)batch * KTOT + kd) * NOUT + n];
      } else { // 2
        v = W[((size_t)batch * NOUT + n) * KTOT + kd];
      }
    }
    Bs[nn][kk] = f2bf(v);
  }
}

template<int KTAPS, int CIN, int BMODE>
__global__ __launch_bounds__(128)
void wmma_gemm(const float* __restrict__ A, const float* __restrict__ W,
               float* __restrict__ Out, int L,
               const float* __restrict__ bias,
               const float* __restrict__ bnscale, const float* __restrict__ bnshift,
               int relu, const unsigned char* __restrict__ rowmask,
               const float* __restrict__ resid, float out_scale)
{
  constexpr int KTOT   = KTAPS * CIN;
  constexpr int KSTEPS = (KTOT + TK - 1) / TK;

  __shared__ unsigned short As[2][TM][TK + 4];
  __shared__ unsigned short Bs[2][TN][TK + 4];

  const int tid  = threadIdx.x;
  const int lane = tid & 31;
  const int wave = tid >> 5;
  const int wm   = (wave >> 1) * 32;
  const int wn   = (wave & 1) * 32;
  const int m0   = blockIdx.y * TM;
  const int n0   = blockIdx.x * TN;
  const int batch = m0 / L;                 // L multiple of 64 -> tile in one batch
  const int pos0  = m0 - batch * L;

  v8f acc[2][2];
  #pragma unroll
  for (int i = 0; i < 2; ++i)
    #pragma unroll
    for (int j = 0; j < 2; ++j)
      #pragma unroll
      for (int r = 0; r < 8; ++r) acc[i][j][r] = 0.0f;

  // prologue: stage k-step 0 into buffer 0
  stage_tiles<KTAPS, CIN, BMODE>(As[0], Bs[0], A, W, L, batch, pos0, n0,
                                 0, tid, KSTEPS > 1);

  const int h    = (lane >> 4) & 1;
  const int rsel = lane & 15;

  for (int ks = 0; ks < KSTEPS; ++ks) {
    __syncthreads();   // buffer[ks&1] staged; previous compute (ks-1) finished
    if (ks + 1 < KSTEPS)
      stage_tiles<KTAPS, CIN, BMODE>(As[(ks + 1) & 1], Bs[(ks + 1) & 1],
                                     A, W, L, batch, pos0, n0,
                                     (ks + 1) * TK, tid, ks + 2 < KSTEPS);

    // ---- fragments from buffer[ks&1] (documented 16-bit A/B K-packing) ----
    const int cb = ks & 1;
    union Frag { v16bf v; unsigned u[8]; } fa[2], fb[2];
    #pragma unroll
    for (int mi = 0; mi < 2; ++mi) {
      const unsigned short* rowp = &As[cb][wm + mi * 16 + rsel][0];
      #pragma unroll
      for (int j = 0; j < 8; ++j) {
        int kb = (j < 4) ? (2 * j + 8 * h) : (16 + 2 * (j - 4) + 8 * h);
        fa[mi].u[j] = *reinterpret_cast<const unsigned*>(rowp + kb);
      }
    }
    #pragma unroll
    for (int ni = 0; ni < 2; ++ni) {
      const unsigned short* rowp = &Bs[cb][wn + ni * 16 + rsel][0];
      #pragma unroll
      for (int j = 0; j < 8; ++j) {
        int kb = (j < 4) ? (2 * j + 8 * h) : (16 + 2 * (j - 4) + 8 * h);
        fb[ni].u[j] = *reinterpret_cast<const unsigned*>(rowp + kb);
      }
    }
    #pragma unroll
    for (int mi = 0; mi < 2; ++mi)
      #pragma unroll
      for (int ni = 0; ni < 2; ++ni)
        acc[mi][ni] = __builtin_amdgcn_wmma_f32_16x16x32_bf16(
            false, fa[mi].v, false, fb[ni].v, (short)0, acc[mi][ni], false, false);
  }

  // ---- fused epilogue: scale, +bias, BN, ReLU, mask, +residual ----
  const int csel = lane & 15;
  #pragma unroll
  for (int mi = 0; mi < 2; ++mi) {
    #pragma unroll
    for (int ni = 0; ni < 2; ++ni) {
      #pragma unroll
      for (int r = 0; r < 8; ++r) {
        int row = m0 + wm + mi * 16 + r + 8 * h;
        int col = n0 + wn + ni * 16 + csel;
        float v = acc[mi][ni][r] * out_scale;
        if (bias)    v += bias[col];
        if (bnscale) v = v * bnscale[col] + bnshift[col];
        if (relu)    v = fmaxf(v, 0.0f);
        if (rowmask && rowmask[row]) v = 0.0f;
        size_t o = (size_t)row * NOUT + col;
        if (resid) v += resid[o];
        Out[o] = v;
      }
    }
  }
}

// ---------------------------------------------------------------------------
// Small fused kernels
// ---------------------------------------------------------------------------
__global__ void k_bnfold(const float* g, const float* be, const float* m,
                         const float* v, float* scale, float* shift, int C) {
  int i = blockIdx.x * blockDim.x + threadIdx.x;
  if (i < C) {
    float s = g[i] * rsqrtf(v[i] + BN_EPS);
    scale[i] = s;
    shift[i] = be[i] - m[i] * s;
  }
}

__global__ void k_frame_sum(const float* __restrict__ afeat, float* __restrict__ fs,
                            int nframes) {
  int f = blockIdx.x * blockDim.x + threadIdx.x;
  if (f >= nframes) return;
  const float* p = afeat + (size_t)f * HID_;
  float s = 0.0f;
  for (int i = 0; i < HID_; ++i) s += p[i];
  fs[f] = s;
}

// per (bh): thread t owns text segment t; ragged scalar mean
__global__ void k_segmean(const int* __restrict__ align, const float* __restrict__ fs,
                          float* __restrict__ aud_scalar) {
  int b = blockIdx.x, t = threadIdx.x;
  int start = 0;
  for (int j = 0; j < t; ++j) start += align[b * TL_ + j];
  int len = align[b * TL_ + t];
  int s0 = start < AL_ ? (start < 0 ? 0 : start) : AL_;
  int e0 = start + len; if (e0 > AL_) e0 = AL_;
  float s = 0.0f;
  const float* f = fs + (size_t)b * AL_;
  for (int p = s0; p < e0; ++p) s += f[p];
  int cl = len > 1 ? len : 1;
  aud_scalar[b * TL_ + t] = s / ((float)cl * (float)HID_);
}

__global__ void k_broadcast_aud(const float* __restrict__ sc, float* __restrict__ aud) {
  size_t i = (size_t)blockIdx.x * blockDim.x + threadIdx.x;
  if (i < (size_t)BH * TL_ * HID_) aud[i] = sc[i / HID_];
}

// row-softmax of s (+key bias from tmask); trans=1 reads s transposed
__global__ __launch_bounds__(256)
void k_softmax(const float* __restrict__ s, const unsigned char* __restrict__ tmask,
               float* __restrict__ P, int trans) {
  int row = blockIdx.x;
  int b = row / TL_, q = row % TL_;
  int c = threadIdx.x;
  __shared__ float red[256];
  float x = trans ? s[((size_t)b * TL_ + c) * TL_ + q]
                  : s[((size_t)b * TL_ + q) * TL_ + c];
  if (tmask[b * TL_ + c]) x += -1e9f;
  red[c] = x; __syncthreads();
  for (int st = 128; st > 0; st >>= 1) { if (c < st) red[c] = fmaxf(red[c], red[c + st]); __syncthreads(); }
  float mx = red[0]; __syncthreads();
  float e = __expf(x - mx);
  red[c] = e; __syncthreads();
  for (int st = 128; st > 0; st >>= 1) { if (c < st) red[c] += red[c + st]; __syncthreads(); }
  P[(size_t)row * TL_ + c] = e / red[0];
}

// prev += LN(xin + y2)*g + be   (in-place update of prev)
__global__ __launch_bounds__(256)
void k_ln_res(const float* __restrict__ xin, const float* __restrict__ y2,
              const float* __restrict__ g, const float* __restrict__ be,
              float* __restrict__ prev) {
  int row = blockIdx.x, c = threadIdx.x;
  __shared__ float red[256];
  size_t o = (size_t)row * HID_ + c;
  float x = xin[o] + y2[o];
  red[c] = x; __syncthreads();
  for (int st = 128; st > 0; st >>= 1) { if (c < st) red[c] += red[c + st]; __syncthreads(); }
  float mu = red[0] * (1.0f / HID_); __syncthreads();
  float d = x - mu;
  red[c] = d * d; __syncthreads();
  for (int st = 128; st > 0; st >>= 1) { if (c < st) red[c] += red[c + st]; __syncthreads(); }
  float var = red[0] * (1.0f / HID_);
  prev[o] = prev[o] + d * rsqrtf(var + 1e-5f) * g[c] + be[c];
}

__global__ void k_concat(const float* __restrict__ txt, const float* __restrict__ aud,
                         float* __restrict__ cat) {
  size_t i = (size_t)blockIdx.x * blockDim.x + threadIdx.x;
  if (i >= (size_t)BH * TL_ * 2 * HID_) return;
  size_t row = i / (2 * HID_);
  int c = (int)(i % (2 * HID_));
  cat[i] = (c < HID_) ? txt[row * HID_ + c] : aud[row * HID_ + (c - HID_)];
}

// alpha = sigmoid(h1 . a2_w + a2_b); out = alpha*txt + (1-alpha)*aud
__global__ __launch_bounds__(256)
void k_alpha(const float* __restrict__ h1, const float* __restrict__ a2w,
             const float* __restrict__ a2b, const float* __restrict__ txt,
             const float* __restrict__ aud, float* __restrict__ out) {
  int row = blockIdx.x, c = threadIdx.x;
  __shared__ float red[256];
  size_t o = (size_t)row * HID_ + c;
  red[c] = h1[o] * a2w[c]; __syncthreads();
  for (int st = 128; st > 0; st >>= 1) { if (c < st) red[c] += red[c + st]; __syncthreads(); }
  float z = red[0] + a2b[0];
  float a = 1.0f / (1.0f + __expf(-z));
  out[o] = a * txt[o] + (1.0f - a) * aud[o];
}

// ---------------------------------------------------------------------------
// Typed GEMM launcher
// ---------------------------------------------------------------------------
template<int KT, int CI, int BM>
static inline void rungemm(const float* A, const float* W, float* Out, int L,
                           const float* bias, const float* bsc, const float* bsh,
                           int relu, const unsigned char* msk, const float* res,
                           float osc, hipStream_t stream) {
  int M = BH * L;
  dim3 g(NOUT / TN, M / TM), b(128);
  hipLaunchKernelGGL((wmma_gemm<KT, CI, BM>), g, b, 0, stream,
                     A, W, Out, L, bias, bsc, bsh, relu, msk, res, osc);
}

// ---------------------------------------------------------------------------
// Host-side orchestration
// ---------------------------------------------------------------------------
extern "C" void kernel_launch(void* const* d_in, const int* in_sizes, int n_in,
                              void* d_out, int out_size, void* d_ws, size_t ws_size,
                              hipStream_t stream) {
  (void)in_sizes; (void)n_in; (void)out_size; (void)ws_size;

  // -------- top-level inputs (dict insertion order) --------
  const float*         text  = (const float*)d_in[0];  // (bh, TL, TDIM)
  const float*         audio = (const float*)d_in[1];  // (bh, AL, NMELS)
  /* d_in[2] = history_mask (unused by reference forward) */
  const unsigned char* tmask = (const unsigned char*)d_in[3]; // (bh, TL) bool
  const unsigned char* amask = (const unsigned char*)d_in[4]; // (bh, AL) bool
  const int*           align = (const int*)d_in[5];           // (bh, TL) i32

  // -------- params: jax pytree flatten (sorted dict keys; list/tuple order)
  int p = 6;
  const float* a1_b = (const float*)d_in[p++];
  const float* a1_w = (const float*)d_in[p++];      // (512,256) [K,N]
  const float* a2_b = (const float*)d_in[p++];
  const float* a2_w = (const float*)d_in[p++];      // (256,1)
  const float* ab_b1 = (const float*)d_in[p++];
  const float* ab_b2 = (const float*)d_in[p++];
  const float* bn1_g = (const float*)d_in[p++];
  const float* bn1_be = (const float*)d_in[p++];
  const float* bn1_m = (const float*)d_in[p++];
  const float* bn1_v = (const float*)d_in[p++];
  const float* bn2_g = (const float*)d_in[p++];
  const float* bn2_be = (const float*)d_in[p++];
  const float* bn2_m = (const float*)d_in[p++];
  const float* bn2_v = (const float*)d_in[p++];
  const float* ab_w1 = (const float*)d_in[p++];     // (256,256,3)
  const float* ab_w2 = (const float*)d_in[p++];     // (256,256,3)
  const float* aout_b = (const float*)d_in[p++];
  const float* aout_w = (const float*)d_in[p++];    // (256,256) [K,N]
  struct Layer {
    const float *a_b1,*a_b2,*a_be,*a_g,*a_w1,*a_w2;
    const float *t_b1,*t_b2,*t_be,*t_g,*t_w1,*t_w2;
    const float *wk,*wq,*wva,*wvt;
  } ly[2];
  for (int l = 0; l < 2; ++l) {
    ly[l].a_b1 = (const float*)d_in[p++];
    ly[l].a_b2 = (const float*)d_in[p++];
    ly[l].a_be = (const float*)d_in[p++];
    ly[l].a_g  = (const float*)d_in[p++];
    ly[l].a_w1 = (const float*)d_in[p++];   // (256,256,9)
    ly[l].a_w2 = (const float*)d_in[p++];   // (256,256,1)
    ly[l].t_b1 = (const float*)d_in[p++];
    ly[l].t_b2 = (const float*)d_in[p++];
    ly[l].t_be = (const float*)d_in[p++];
    ly[l].t_g  = (const float*)d_in[p++];
    ly[l].t_w1 = (const float*)d_in[p++];
    ly[l].t_w2 = (const float*)d_in[p++];
    ly[l].wk   = (const float*)d_in[p++];   // (256,256) [K,N]
    ly[l].wq   = (const float*)d_in[p++];
    ly[l].wva  = (const float*)d_in[p++];
    ly[l].wvt  = (const float*)d_in[p++];
  }
  const float* prenet_b = (const float*)d_in[p++];
  const float* pre_g  = (const float*)d_in[p++];
  const float* pre_be = (const float*)d_in[p++];
  const float* pre_m  = (const float*)d_in[p++];
  const float* pre_v  = (const float*)d_in[p++];
  const float* prenet_w = (const float*)d_in[p++];  // (256,80,3)
  const float* text_b = (const float*)d_in[p++];
  const float* text_w = (const float*)d_in[p++];    // (512,256) [K,N]

  // -------- workspace layout (floats) --------
  float* ws = (float*)d_ws;
  const size_t SBIG = (size_t)BH * AL_ * HID_;   // 33.5M floats
  const size_t SSM  = (size_t)BH * TL_ * HID_;   // 4.19M floats
  float* bufA = ws;                 // encoder ping
  float* bufB = ws + SBIG;          // encoder pong / afeat
  // per-layer temporaries reuse the (then dead) encoder region:
  float* qb  = ws + 0 * SSM;  float* kb  = ws + 1 * SSM;
  float* vab = ws + 2 * SSM;  float* vtb = ws + 3 * SSM;
  float* sb  = ws + 4 * SSM;  float* Pt  = ws + 5 * SSM;
  float* Pa  = ws + 6 * SSM;  float* tob = ws + 7 * SSM;
  float* aob = ws + 8 * SSM;  float* pwy = ws + 9 * SSM;
  float* y2b = ws + 10 * SSM;
  float* txt = ws + 2 * SBIG;
  float* aud = txt + SSM;
  float* cat = aud + SSM;
  float* h1  = cat + 2 * SSM;
  float* fs  = h1 + SSM;                          // BH*AL
  float* ascal = fs + (size_t)BH * AL_;           // BH*TL
  float* bnp = ascal + (size_t)BH * TL_;          // 6*HID folded BN params
  float *bnPs = bnp, *bnPb = bnp + HID_;
  float *bn1s = bnp + 2 * HID_, *bn1b = bnp + 3 * HID_;
  float *bn2s = bnp + 4 * HID_, *bn2b = bnp + 5 * HID_;

  // fold BN params
  k_bnfold<<<1, 256, 0, stream>>>(pre_g, pre_be, pre_m, pre_v, bnPs, bnPb, HID_);
  k_bnfold<<<1, 256, 0, stream>>>(bn1_g, bn1_be, bn1_m, bn1_v, bn1s, bn1b, HID_);
  k_bnfold<<<1, 256, 0, stream>>>(bn2_g, bn2_be, bn2_m, bn2_v, bn2s, bn2b, HID_);

  // 1) text projection + mask
  rungemm<1, TDIM_, 3>(text, text_w, txt, TL_, text_b, nullptr, nullptr, 0, tmask, nullptr, 1.0f, stream);

  // 2) audio prenet conv(K=3) + BN + relu + mask
  rungemm<3, NMELS_, 0>(audio, prenet_w, bufA, AL_, prenet_b, bnPs, bnPb, 1, amask, nullptr, 1.0f, stream);
  // 3) residual block
  rungemm<3, HID_, 0>(bufA, ab_w1, bufB, AL_, ab_b1, bn1s, bn1b, 1, nullptr, nullptr, 1.0f, stream);
  rungemm<3, HID_, 0>(bufB, ab_w2, bufA, AL_, ab_b2, bn2s, bn2b, 1, amask, bufA, 1.0f, stream);
  // 4) aout projection + mask -> afeat in bufB
  rungemm<1, HID_, 3>(bufA, aout_w, bufB, AL_, aout_b, nullptr, nullptr, 0, amask, nullptr, 1.0f, stream);

  // 5) ragged segment means -> aud
  k_frame_sum<<<(BH * AL_ + 255) / 256, 256, 0, stream>>>(bufB, fs, BH * AL_);
  k_segmean<<<BH, TL_, 0, stream>>>(align, fs, ascal);
  k_broadcast_aud<<<(int)((SSM + 255) / 256), 256, 0, stream>>>(ascal, aud);

  // 6) bidirectional attention + conv-FFN layers
  for (int l = 0; l < 2; ++l) {
    rungemm<1, HID_, 3>(txt, ly[l].wq,  qb,  TL_, nullptr, nullptr, nullptr, 0, nullptr, nullptr, 1.0f, stream);
    rungemm<1, HID_, 3>(aud, ly[l].wk,  kb,  TL_, nullptr, nullptr, nullptr, 0, nullptr, nullptr, 1.0f, stream);
    rungemm<1, HID_, 3>(aud, ly[l].wva, vab, TL_, nullptr, nullptr, nullptr, 0, nullptr, nullptr, 1.0f, stream);
    rungemm<1, HID_, 3>(txt, ly[l].wvt, vtb, TL_, nullptr, nullptr, nullptr, 0, nullptr, nullptr, 1.0f, stream);
    // s = (q @ k^T) / sqrt(HID)   (batched, B transposed)
    rungemm<1, HID_, 2>(qb, kb, sb, TL_, nullptr, nullptr, nullptr, 0, nullptr, nullptr, 0.0625f, stream);
    k_softmax<<<BH * TL_, 256, 0, stream>>>(sb, tmask, Pt, 0);
    k_softmax<<<BH * TL_, 256, 0, stream>>>(sb, tmask, Pa, 1);
    // attn outputs (batched, normal) + mask
    rungemm<1, TL_, 1>(Pt, vab, tob, TL_, nullptr, nullptr, nullptr, 0, tmask, nullptr, 1.0f, stream);
    rungemm<1, TL_, 1>(Pa, vtb, aob, TL_, nullptr, nullptr, nullptr, 0, tmask, nullptr, 1.0f, stream);
    // audio pwff: conv(K=9)+relu, conv(K=1), LN residual into aud
    rungemm<9, HID_, 0>(aob, ly[l].a_w1, pwy, TL_, ly[l].a_b1, nullptr, nullptr, 1, nullptr, nullptr, 1.0f, stream);
    rungemm<1, HID_, 0>(pwy, ly[l].a_w2, y2b, TL_, ly[l].a_b2, nullptr, nullptr, 0, nullptr, nullptr, 1.0f, stream);
    k_ln_res<<<BH * TL_, 256, 0, stream>>>(aob, y2b, ly[l].a_g, ly[l].a_be, aud);
    // text pwff
    rungemm<9, HID_, 0>(tob, ly[l].t_w1, pwy, TL_, ly[l].t_b1, nullptr, nullptr, 1, nullptr, nullptr, 1.0f, stream);
    rungemm<1, HID_, 0>(pwy, ly[l].t_w2, y2b, TL_, ly[l].t_b2, nullptr, nullptr, 0, nullptr, nullptr, 1.0f, stream);
    k_ln_res<<<BH * TL_, 256, 0, stream>>>(tob, y2b, ly[l].t_g, ly[l].t_be, txt);
  }

  // 7) gated fusion
  k_concat<<<(int)((2 * SSM + 255) / 256), 256, 0, stream>>>(txt, aud, cat);
  rungemm<1, 2 * HID_, 3>(cat, a1_w, h1, TL_, a1_b, nullptr, nullptr, 1, nullptr, nullptr, 1.0f, stream);
  k_alpha<<<BH * TL_, 256, 0, stream>>>(h1, a2_w, a2_b, txt, aud, (float*)d_out);
}
